// SSMRTF_model_24292335026631
// MI455X (gfx1250) — matrix operand order repaired and checked by reading, not compiled, not run on hardware
//
#include <hip/hip_runtime.h>
#include <math.h>

typedef __attribute__((ext_vector_type(16))) _Float16 v16h;
typedef __attribute__((ext_vector_type(8)))  float    v8f;

#define BSZ   8
#define LSEQ  4096
#define CCHAN 1024
#define NTAP  64
#define TT    128           // time steps per workgroup
#define CCH   32            // channels per workgroup
#define XT    (TT + NTAP)   // staged x window (192)
#define XSTR  202           // padded LDS stride (halfs); 101 dwords, odd

// ---------------------------------------------------------------------------
// Pre-kernel: truncated impulse response of B(z)/A(z) (= shift(B) when A==0),
// stored as f16 taps[c][j] meaning  y[t] += taps[c][j] * x[t-1-j].
// ---------------------------------------------------------------------------
__global__ __launch_bounds__(64)
void rtf_taps_kernel(const float* __restrict__ Acoef,
                     const float* __restrict__ Bcoef,
                     _Float16* __restrict__ taps)
{
    __shared__ float k[64][NTAP + 1];
    const int c = blockIdx.x * 64 + threadIdx.x;
    float* kc = k[threadIdx.x];
    kc[0] = 0.f;
    for (int t = 1; t <= NTAP; ++t) {
        float s = Bcoef[(size_t)c * NTAP + (t - 1)];
        for (int i = 0; i <= t - 2; ++i)
            s -= Acoef[(size_t)c * NTAP + i] * kc[t - 1 - i];
        kc[t] = s;
    }
    for (int j = 0; j < NTAP; ++j)
        taps[(size_t)c * NTAP + j] = (_Float16)kc[j + 1];
}

// ---------------------------------------------------------------------------
// Main kernel: depthwise causal FIR via V_WMMA_F32_16X16X32_F16 + exact GELU.
// x window staged TIME-REVERSED in LDS so every B-fragment is an ascending
// contiguous 32-byte run -> raw ds_load_b128 pairs, no half-swizzle VALU.
// ---------------------------------------------------------------------------
__global__ __launch_bounds__(256)
void ssm_rtf_fir_wmma(const float* __restrict__ x,
                      const float* __restrict__ Acoef,
                      const float* __restrict__ Bcoef,
                      const float* __restrict__ h0,
                      const _Float16* __restrict__ taps,   // may be null
                      float* __restrict__ out)
{
    __shared__ _Float16 coefH[CCH][NTAP];      // 4 KB
    __shared__ _Float16 xr[CCH][XSTR];         // ~12.6 KB, [ch][reversed time]
    __shared__ float    outT[TT][CCH + 1];     // 16.9 KB, padded

    const int tid = threadIdx.x;
    const int c0  = blockIdx.x * CCH;
    const int t0  = blockIdx.y * TT;
    const int b   = blockIdx.z;

    // ---- Phase A: tap coefficients into LDS (f16) ----
    if (taps != nullptr) {
        for (int e = tid; e < CCH * NTAP; e += 256)
            coefH[e >> 6][e & 63] = taps[(size_t)(c0 + (e >> 6)) * NTAP + (e & 63)];
    } else if (tid < CCH) {
        // in-block IIR recursion; scratch aliases outT (not yet used)
        float* kc = ((float*)outT) + tid * (NTAP + 1);
        const int c = c0 + tid;
        kc[0] = 0.f;
        for (int t = 1; t <= NTAP; ++t) {
            float s = Bcoef[(size_t)c * NTAP + (t - 1)];
            for (int i = 0; i <= t - 2; ++i)
                s -= Acoef[(size_t)c * NTAP + i] * kc[t - 1 - i];
            kc[t] = s;
        }
        for (int j = 0; j < NTAP; ++j) coefH[tid][j] = (_Float16)kc[j + 1];
    }

    // ---- Phase B: stage x[t0-64 .. t0+127] x 32ch as f16, time-REVERSED ----
    // xr[cl][r] = x[b, t0+127-r, c0+cl]   (r = 0..191)
    for (int e = tid; e < XT * CCH; e += 256) {
        const int tl = e >> 5;              // 0..191
        const int cl = e & 31;              // coalesced: lane -> channel
        const int t  = t0 - NTAP + tl;
        float v = (t >= 0) ? x[((size_t)b * LSEQ + t) * CCHAN + c0 + cl] : 0.f;
        xr[cl][(XT - 1) - tl] = (_Float16)v;
    }
    __syncthreads();

    // ---- Phase C: WMMA FIR. Wave owns 4 channels, loops 8 time blocks. ----
    const int wid    = tid >> 5;
    const int lane   = tid & 31;
    const int laneHi = lane >> 4;           // 0: lanes 0-15, 1: lanes 16-31
    const int n      = lane & 15;           // output-time column / B column

    // A operand = taps replicated across M. K order matches B fragments:
    //   half i : K = 16*laneHi + i
    v16h afrag[4][2];
    for (int cc = 0; cc < 4; ++cc) {
        const int cl = wid * 4 + cc;
        for (int kb = 0; kb < 2; ++kb) {
            v16h a;
#pragma unroll
            for (int i = 0; i < 16; ++i) {
                const int K = laneHi * 16 + i;
                a[i] = coefH[cl][kb * 32 + K];
            }
            afrag[cc][kb] = a;
        }
    }

    for (int tt = 0; tt < 8; ++tt) {
        const int tb = tt * 16;                       // time base rel. t0
        // B half i (K = 16*laneHi + i + 32*kb) = x[t0+tb+n-1-K]
        //   = xr[cl][r0 + 32*kb + i],  r0 = 128 - tb - n + 16*laneHi
        const int r0 = 128 - tb - n + 16 * laneHi;
#pragma unroll
        for (int cp = 0; cp < 2; ++cp) {
            const int cl0 = wid * 4 + 2 * cp;
            const int cl1 = cl0 + 1;
            v16h b0k0, b0k1, b1k0, b1k1;
#pragma unroll
            for (int i = 0; i < 16; ++i) {
                b0k0[i] = xr[cl0][r0 + i];
                b1k0[i] = xr[cl1][r0 + i];
            }
#pragma unroll
            for (int i = 0; i < 16; ++i) {
                b0k1[i] = xr[cl0][r0 + 32 + i];
                b1k1[i] = xr[cl1][r0 + 32 + i];
            }
            v8f a0 = {}, a1 = {};
            // two independent accumulator chains interleaved: the other
            // channel's WMMA fills each chain's D->C hazard window
            a0 = __builtin_amdgcn_wmma_f32_16x16x32_f16(false, afrag[2*cp  ][0], false, b0k0, (short)0, a0, false, false);
            a1 = __builtin_amdgcn_wmma_f32_16x16x32_f16(false, afrag[2*cp+1][0], false, b1k0, (short)0, a1, false, false);
            a0 = __builtin_amdgcn_wmma_f32_16x16x32_f16(false, afrag[2*cp  ][1], false, b0k1, (short)0, a0, false, false);
            a1 = __builtin_amdgcn_wmma_f32_16x16x32_f16(false, afrag[2*cp+1][1], false, b1k1, (short)0, a1, false, false);
            // all M rows of D identical; column n carries y[t0+tb+n].
            // both lane halves write the same value to the same address.
            outT[tb + n][cl0] = a0[0];
            outT[tb + n][cl1] = a1[0];
        }
    }
    __syncthreads();

    // ---- Phase D: skip + exact-erf GELU, coalesced store ----
    const float hh = h0[0];
    for (int e = tid; e < TT * CCH; e += 256) {
        const int tl = e >> 5;
        const int cl = e & 31;
        const size_t gidx = ((size_t)b * LSEQ + (t0 + tl)) * CCHAN + c0 + cl;
        const float y = outT[tl][cl] + hh * x[gidx];
        out[gidx] = 0.5f * y * (1.0f + erff(y * 0.7071067811865475f));
    }
}

// ---------------------------------------------------------------------------
extern "C" void kernel_launch(void* const* d_in, const int* in_sizes, int n_in,
                              void* d_out, int out_size, void* d_ws, size_t ws_size,
                              hipStream_t stream)
{
    const float* x  = (const float*)d_in[0];
    const float* A  = (const float*)d_in[1];
    const float* B  = (const float*)d_in[2];
    const float* h0 = (const float*)d_in[3];
    float* out = (float*)d_out;

    const size_t tapsBytes = (size_t)CCHAN * NTAP * sizeof(_Float16); // 128 KB
    _Float16* taps = nullptr;
    if (d_ws && ws_size >= tapsBytes) {
        taps = (_Float16*)d_ws;
        rtf_taps_kernel<<<CCHAN / 64, 64, 0, stream>>>(A, B, taps);
    }

    dim3 grid(CCHAN / CCH, LSEQ / TT, BSZ);   // (32, 32, 8) workgroups
    ssm_rtf_fir_wmma<<<grid, 256, 0, stream>>>(x, A, B, h0, taps, out);
}
